// Self_Attention_14491219657355
// MI455X (gfx1250) — matrix operand order repaired and checked by reading, not compiled, not run on hardware
//
#include <hip/hip_runtime.h>
#include <cstdint>

typedef _Float16 v16h __attribute__((ext_vector_type(16)));
typedef _Float16 v8h  __attribute__((ext_vector_type(8)));
typedef float    v8f  __attribute__((ext_vector_type(8)));

#define B_   2
#define S_   2048
#define D_   1024
#define H_   8
#define HD_  128
#define TOPK 32
#define ROWS (B_*H_*S_)   /* 32768 */

#define L2E 1.44269504088896340736f

__device__ __forceinline__ float exp2_fast(float x){ return __builtin_amdgcn_exp2f(x); }

union F16x16 { v16h v; v8h h[2]; };

__device__ __forceinline__ v8h cvt8(const float4& a, const float4& b){
  v8h r;
  r[0]=(_Float16)a.x; r[1]=(_Float16)a.y; r[2]=(_Float16)a.z; r[3]=(_Float16)a.w;
  r[4]=(_Float16)b.x; r[5]=(_Float16)b.y; r[6]=(_Float16)b.z; r[7]=(_Float16)b.w;
  return r;
}
// load 8 f32, scale, convert to 8 f16 (A-fragment group)
__device__ __forceinline__ v8h load8hs(const float* p, float sc){
  const float4* p4 = reinterpret_cast<const float4*>(p);
  float4 a = p4[0], b = p4[1];
  v8h r;
  r[0]=(_Float16)(a.x*sc); r[1]=(_Float16)(a.y*sc); r[2]=(_Float16)(a.z*sc); r[3]=(_Float16)(a.w*sc);
  r[4]=(_Float16)(b.x*sc); r[5]=(_Float16)(b.y*sc); r[6]=(_Float16)(b.z*sc); r[7]=(_Float16)(b.w*sc);
  return r;
}

// ---------------------------------------------------------------------------
// Kernel 1: exact top-32 per attention_weight row via radix-select on
// 43-bit keys ((order-transformed f32)<<11 | (2047-pos)) -> stable-argsort
// semantics (ties broken by smaller position); unique keys so mask = key>=T.
// ---------------------------------------------------------------------------
__global__ __launch_bounds__(256) void topk_kernel(const float* __restrict__ aw,
                                                   int* __restrict__ idx_out){
  __shared__ uint32_t su[S_];
  __shared__ uint32_t hist[256];
  __shared__ int sh_sel, sh_k, sh_cnt;
  const int tid = threadIdx.x;
  const long long row = blockIdx.x;
  const float* rp = aw + row * (long long)S_;

  #pragma unroll
  for (int i=0;i<8;i++){
    int p = i*256 + tid;
    uint32_t bts = __float_as_uint(rp[p]);
    su[p] = bts ^ ((uint32_t)(((int32_t)bts) >> 31) | 0x80000000u); // monotone map
  }
  if (tid==0) sh_cnt = 0;
  __syncthreads();

  unsigned long long prefix = 0ull, pmask = 0ull;
  int k = TOPK;
  for (int pass=0; pass<6; pass++){
    const int shift = 40 - 8*pass;
    hist[tid] = 0;
    __syncthreads();
    #pragma unroll
    for (int i=0;i<8;i++){
      int p = i*256 + tid;
      unsigned long long key = ((unsigned long long)su[p] << 11) |
                               (unsigned long long)(2047 - p);
      if ((key & pmask) == prefix)
        atomicAdd(&hist[(unsigned)(key>>shift) & 255u], 1u);
    }
    __syncthreads();
    uint32_t own = hist[tid];
    // inclusive suffix scan (Hillis-Steele, in place)
    for (int st=1; st<256; st<<=1){
      uint32_t t = (tid+st < 256) ? hist[tid+st] : 0u;
      __syncthreads();
      hist[tid] += t;
      __syncthreads();
    }
    uint32_t incl = hist[tid];
    uint32_t excl = incl - own;
    if ((int)excl < k && (int)incl >= k){ sh_sel = tid; sh_k = k - (int)excl; }
    __syncthreads();
    prefix |= ((unsigned long long)sh_sel) << shift;
    pmask  |= (255ull << shift);
    k = sh_k;
    __syncthreads();
  }
  // gather indices of the 32 keys >= exact threshold
  #pragma unroll
  for (int i=0;i<8;i++){
    int p = i*256 + tid;
    unsigned long long key = ((unsigned long long)su[p] << 11) |
                             (unsigned long long)(2047 - p);
    if (key >= prefix){
      int slot = atomicAdd(&sh_cnt, 1);
      if (slot < TOPK) idx_out[row*TOPK + slot] = p;
    }
  }
}

// ---------------------------------------------------------------------------
// Kernel 2a: streaming QK^T via V_WMMA_F32_16X16X32_F16, producing only the
// per-row softmax stats (M, Z) online.  One wave = 32 queries (two A-tiles)
// vs all 2048 keys; 8 waves/block share the head's k-tiles through a
// double-buffered f16 LDS stage (single barrier per tile, global loads for
// tile kb+1 in flight under the 8-WMMA chain of tile kb).  Q is pre-scaled
// by log2(e) so every softmax exp is a bare v_exp_f32 (2^x).
// ---------------------------------------------------------------------------
__global__ __launch_bounds__(256) void maxz_kernel(const float* __restrict__ q,
                                                   const float* __restrict__ kmat,
                                                   float* __restrict__ Mout,
                                                   float* __restrict__ Zout){
  __shared__ __align__(16) _Float16 kh[2][16*HD_];   // 2 x 4 KB k-tile (f16)
  const int tid  = threadIdx.x;
  const int wave = tid >> 5;
  const int lane = tid & 31;
  const int n16  = lane & 15;
  const int half = lane >> 4;

  const int hd = blockIdx.x >> 3;        // 0..15 == b*8+h
  const int qg = blockIdx.x & 7;
  const int b  = hd >> 3;
  const int h  = hd & 7;
  const int qbase = (qg*8 + wave) * 32;  // this wave's 32 queries

  // A fragments for two 16-row query tiles (ISA 7.12.2 A layout), q * log2e
  F16x16 a0[4], a1[4];
  {
    const float* qr0 = q + ((size_t)(b*S_ + qbase + n16))*D_ + h*HD_;
    const float* qr1 = qr0 + (size_t)16*D_;
    #pragma unroll
    for (int c=0;c<4;c++){
      a0[c].h[0] = load8hs(qr0 + c*32 +      8*half, L2E);
      a0[c].h[1] = load8hs(qr0 + c*32 + 16 + 8*half, L2E);
      a1[c].h[0] = load8hs(qr1 + c*32 +      8*half, L2E);
      a1[c].h[1] = load8hs(qr1 + c*32 + 16 + 8*half, L2E);
    }
  }

  float m0[8], z0[8], m1[8], z1[8];
  #pragma unroll
  for (int r=0;r<8;r++){ m0[r]=-3.4e38f; z0[r]=0.f; m1[r]=-3.4e38f; z1[r]=0.f; }

  const int srow = tid >> 4;
  const int scol = (tid & 15) * 8;
  const float* kbp = kmat + ((size_t)(b*S_))*D_ + h*HD_ + (size_t)srow*D_ + scol;

  // prefetch tile 0 into registers
  float4 pf0, pf1;
  {
    const float4* kp4 = reinterpret_cast<const float4*>(kbp);
    pf0 = kp4[0]; pf1 = kp4[1];
  }

  for (int kb=0; kb<S_/16; kb++){
    // stage prefetched tile kb into buf[kb&1]
    *reinterpret_cast<v8h*>(&kh[kb&1][srow*HD_ + scol]) = cvt8(pf0, pf1);
    // launch global loads for tile kb+1 (consumed next iteration)
    if (kb+1 < S_/16){
      const float4* kp4 = reinterpret_cast<const float4*>(kbp + (size_t)(kb+1)*16*D_);
      pf0 = kp4[0]; pf1 = kp4[1];
    }
    __syncthreads();

    const _Float16* cur = kh[kb&1];
    F16x16 bf[4];                         // B layout: K = 8*half+s, 16+8*half+(s-8)
    #pragma unroll
    for (int c=0;c<4;c++){
      bf[c].h[0] = *reinterpret_cast<const v8h*>(&cur[n16*HD_ + c*32 + 8*half]);
      bf[c].h[1] = *reinterpret_cast<const v8h*>(&cur[n16*HD_ + c*32 + 16 + 8*half]);
    }
    v8f acc0 = {}, acc1 = {};
    #pragma unroll
    for (int c=0;c<4;c++)
      acc0 = __builtin_amdgcn_wmma_f32_16x16x32_f16(false, a0[c].v, false, bf[c].v,
                                                    (short)0, acc0, false, false);
    #pragma unroll
    for (int c=0;c<4;c++)
      acc1 = __builtin_amdgcn_wmma_f32_16x16x32_f16(false, a1[c].v, false, bf[c].v,
                                                    (short)0, acc1, false, false);

    // per-lane online (m,z): lane owns key column n16; 16 query rows
    #pragma unroll
    for (int r=0;r<8;r++){
      {
        float s  = acc0[r];
        float mo = m0[r];
        float mn = fmaxf(mo, s);
        float e  = exp2_fast(fminf(mo, s) - mn);   // 2^(-|s-m|), scores in log2 units
        z0[r] = (s > mo) ? z0[r]*e + 1.0f : z0[r] + e;
        m0[r] = mn;
      }
      {
        float s  = acc1[r];
        float mo = m1[r];
        float mn = fmaxf(mo, s);
        float e  = exp2_fast(fminf(mo, s) - mn);
        z1[r] = (s > mo) ? z1[r]*e + 1.0f : z1[r] + e;
        m1[r] = mn;
      }
    }
  }

  // combine 16 key-columns within each half-wave
  #pragma unroll
  for (int r=0;r<8;r++){
    float mr = m0[r], zr = z0[r];
    #pragma unroll
    for (int off=1; off<16; off<<=1){
      float mo = __shfl_xor(mr, off, 32);
      float zo = __shfl_xor(zr, off, 32);
      float mn = fmaxf(mr, mo);
      zr = zr*exp2_fast(mr-mn) + zo*exp2_fast(mo-mn);
      mr = mn;
    }
    m0[r] = mr; z0[r] = zr;

    mr = m1[r]; zr = z1[r];
    #pragma unroll
    for (int off=1; off<16; off<<=1){
      float mo = __shfl_xor(mr, off, 32);
      float zo = __shfl_xor(zr, off, 32);
      float mn = fmaxf(mr, mo);
      zr = zr*exp2_fast(mr-mn) + zo*exp2_fast(mo-mn);
      mr = mn;
    }
    m1[r] = mr; z1[r] = zr;
  }
  if (n16 == 0){
    const size_t base = (size_t)hd*S_ + qbase + half*8;
    #pragma unroll
    for (int r=0;r<8;r++){ Mout[base+r]    = m0[r]; Zout[base+r]    = z0[r]; }
    #pragma unroll
    for (int r=0;r<8;r++){ Mout[base+16+r] = m1[r]; Zout[base+16+r] = z1[r]; }
  }
}

// ---------------------------------------------------------------------------
// Kernel 2b: per (b,h,q): 32-key gather. Lane i recomputes s_i = (q*log2e).k_i,
// w_i = 2^(s_i - M); out = sum w_i * v_i / (sum w + eps*Z).
// ---------------------------------------------------------------------------
__global__ __launch_bounds__(256) void gather_kernel(const float* __restrict__ q,
      const float* __restrict__ kmat, const float* __restrict__ vmat,
      const int* __restrict__ idx, const float* __restrict__ Min,
      const float* __restrict__ Zin, float* __restrict__ out){
  __shared__ __align__(16) float qs[8][HD_];
  const int tid  = threadIdx.x;
  const int wave = tid >> 5;
  const int lane = tid & 31;
  const int rowid = blockIdx.x*8 + wave;     // 0..32767
  const int b  = rowid >> 14;
  const int h  = (rowid >> 11) & 7;
  const int qi = rowid & 2047;

  const size_t qoff = ((size_t)(b*S_ + qi))*D_ + h*HD_;
  {
    float4 qv = *reinterpret_cast<const float4*>(q + qoff + lane*4);
    qv.x *= L2E; qv.y *= L2E; qv.z *= L2E; qv.w *= L2E;   // log2-domain scores
    *reinterpret_cast<float4*>(&qs[wave][lane*4]) = qv;
  }
  __syncthreads();

  const int ki = idx[(size_t)rowid*TOPK + lane];
  const float* kp = kmat + ((size_t)(b*S_ + ki))*D_ + h*HD_;
  float s = 0.f;
  #pragma unroll
  for (int d0=0; d0<HD_; d0+=4){
    float4 kv = *reinterpret_cast<const float4*>(kp + d0);
    s += qs[wave][d0]*kv.x + qs[wave][d0+1]*kv.y
       + qs[wave][d0+2]*kv.z + qs[wave][d0+3]*kv.w;
  }
  const float M = Min[rowid];
  const float Z = Zin[rowid];
  float w = exp2_fast(s - M);
  float sumw = w;
  #pragma unroll
  for (int off=1; off<32; off<<=1) sumw += __shfl_xor(sumw, off, 32);
  const float inv = 1.0f / (sumw + 1e-5f*Z);

  float a0=0.f, a1=0.f, a2=0.f, a3=0.f;
  #pragma unroll 4
  for (int kk=0; kk<32; kk++){
    float wk = __shfl(w, kk, 32);
    int   kx = __shfl(ki, kk, 32);
    const float* vp = vmat + ((size_t)(b*S_ + kx))*D_ + h*HD_;
    a0 += wk * vp[lane];
    a1 += wk * vp[lane+32];
    a2 += wk * vp[lane+64];
    a3 += wk * vp[lane+96];
  }
  float* op = out + qoff;
  op[lane]    = a0*inv;
  op[lane+32] = a1*inv;
  op[lane+64] = a2*inv;
  op[lane+96] = a3*inv;
}

// ---------------------------------------------------------------------------
// Kernel 3: in-place LayerNorm over D=1024 (population variance).
// ---------------------------------------------------------------------------
__global__ __launch_bounds__(256) void ln_kernel(float* __restrict__ out,
                                                 const float* __restrict__ gamma,
                                                 const float* __restrict__ beta){
  __shared__ float red[256];
  const int tid = threadIdx.x;
  float* rp = out + (size_t)blockIdx.x * D_;
  float4 x = *reinterpret_cast<const float4*>(rp + tid*4);
  float s  = x.x + x.y + x.z + x.w;
  float ss = x.x*x.x + x.y*x.y + x.z*x.z + x.w*x.w;

  red[tid] = s;  __syncthreads();
  for (int st=128; st>0; st>>=1){ if (tid<st) red[tid]+=red[tid+st]; __syncthreads(); }
  const float mean = red[0] * (1.0f/D_);
  __syncthreads();
  red[tid] = ss; __syncthreads();
  for (int st=128; st>0; st>>=1){ if (tid<st) red[tid]+=red[tid+st]; __syncthreads(); }
  const float var  = red[0] * (1.0f/D_) - mean*mean;
  const float rstd = rsqrtf(var + 1e-5f);

  float4 g  = *reinterpret_cast<const float4*>(gamma + tid*4);
  float4 bb = *reinterpret_cast<const float4*>(beta  + tid*4);
  float4 y;
  y.x = (x.x-mean)*rstd*g.x + bb.x;
  y.y = (x.y-mean)*rstd*g.y + bb.y;
  y.z = (x.z-mean)*rstd*g.z + bb.z;
  y.w = (x.w-mean)*rstd*g.w + bb.w;
  *reinterpret_cast<float4*>(rp + tid*4) = y;
}

extern "C" void kernel_launch(void* const* d_in, const int* in_sizes, int n_in,
                              void* d_out, int out_size, void* d_ws, size_t ws_size,
                              hipStream_t stream) {
  (void)in_sizes; (void)n_in; (void)out_size; (void)ws_size;
  const float* q     = (const float*)d_in[0];
  const float* k     = (const float*)d_in[1];
  const float* v     = (const float*)d_in[2];
  const float* aw    = (const float*)d_in[3];
  const float* gamma = (const float*)d_in[4];
  const float* beta  = (const float*)d_in[5];
  float* out = (float*)d_out;

  // workspace: idx[32768*32] int (4 MB) | M[32768] f32 | Z[32768] f32
  int*   idx = (int*)d_ws;
  float* M   = (float*)((char*)d_ws + (size_t)ROWS*TOPK*sizeof(int));
  float* Z   = M + ROWS;

  topk_kernel  <<<ROWS,   256, 0, stream>>>(aw, idx);
  maxz_kernel  <<<128,    256, 0, stream>>>(q, k, M, Z);
  gather_kernel<<<ROWS/8, 256, 0, stream>>>(q, k, v, idx, M, Z, out);
  ln_kernel    <<<B_*S_,  256, 0, stream>>>(out, gamma, beta);
}
